// MultiVeh_GraphAttentionLayer_40218073760394
// MI455X (gfx1250) — matrix operand order repaired and checked by reading, not compiled, not run on hardware
//
#include <hip/hip_runtime.h>

typedef _Float16 half_t;
typedef __attribute__((ext_vector_type(16))) _Float16 v16h;
typedef __attribute__((ext_vector_type(8)))  _Float16 v8h;
typedef __attribute__((ext_vector_type(8)))  float    v8f;
typedef __attribute__((ext_vector_type(4)))  float    v4f;
typedef __attribute__((ext_vector_type(4)))  int      v4i;

#define BB 16
#define NN 1024
#define FD 256
#define MM (BB * NN)          // 16384 flattened rows
#define LRELU_ALPHA 0.2f
#define NEG_BIG (-9.0e15f)

// ---------------------------------------------------------------------------
// Kernel 0a: transpose + convert W -> WT16[n][k] (f16), so GEMM B-fragments
// become single 32-byte contiguous vector loads.
// ---------------------------------------------------------------------------
__global__ __launch_bounds__(256)
void k_wt(const float* __restrict__ W, half_t* __restrict__ WT16) {
  const int n = blockIdx.x;     // 0..255 (output feature)
  const int k = threadIdx.x;    // 0..255 (input feature)
  WT16[(size_t)n * FD + k] = (half_t)W[(size_t)k * FD + n];
}

// ---------------------------------------------------------------------------
// Kernel 0b: wa1[k] = sum_f W[k,f]*a1[f], wa2[k] = sum_f W[k,f]*a2[f]
// (e_i = (h@W)@a1 == h@(W@a1); avoids an extra pass over Wh)
// ---------------------------------------------------------------------------
__global__ __launch_bounds__(256)
void k_wa(const float* __restrict__ W, const float* __restrict__ a,
          float* __restrict__ wa) {
  const int k = threadIdx.x;                 // 0..255
  const float* wr = W + (size_t)k * FD;
  float s1 = 0.f, s2 = 0.f;
  for (int f = 0; f < FD; ++f) {
    const float w = wr[f];
    s1 += w * a[f];
    s2 += w * a[FD + f];
  }
  wa[k]      = s1;
  wa[FD + k] = s2;
}

// ---------------------------------------------------------------------------
// Kernel 1: e_i[m] = h[m,:]·wa1 ; e_j[m] = h[m,:]·wa2  (one wave per row)
// ---------------------------------------------------------------------------
__global__ __launch_bounds__(256)
void k_scores(const float* __restrict__ h, const float* __restrict__ wa,
              float* __restrict__ e_i, float* __restrict__ e_j) {
  const int lane = threadIdx.x & 31;
  const int row  = blockIdx.x * 8 + (threadIdx.x >> 5);   // 16384 rows
  const v4f* hp = reinterpret_cast<const v4f*>(h + (size_t)row * FD + lane * 8);
  const v4f x0 = hp[0], x1 = hp[1];
  float s1 = 0.f, s2 = 0.f;
#pragma unroll
  for (int u = 0; u < 4; ++u) {
    s1 += x0[u] * wa[lane * 8 + u]      + x1[u] * wa[lane * 8 + 4 + u];
    s2 += x0[u] * wa[FD + lane * 8 + u] + x1[u] * wa[FD + lane * 8 + 4 + u];
  }
#pragma unroll
  for (int off = 16; off >= 1; off >>= 1) {
    s1 += __shfl_xor(s1, off, 32);
    s2 += __shfl_xor(s2, off, 32);
  }
  if (lane == 0) { e_i[row] = s1; e_j[row] = s2; }
}

// ---------------------------------------------------------------------------
// Kernel 2: Wh = h @ W via v_wmma_f32_16x16x32_f16, stored TRANSPOSED as f16:
//   WhT[f][m], f in [0,256), m in [0,16384)
// Grid: 1024 blocks (one 16-row M-tile) x 128 threads (4 waves).
// Wave w owns N-tiles 4w..4w+3 (A fragment reused across 4 WMMAs).
// B fragments are single v16h loads from pre-transposed f16 WT16.
// ---------------------------------------------------------------------------
__global__ __launch_bounds__(128)
void k_gemm1(const float* __restrict__ h, const half_t* __restrict__ WT16,
             half_t* __restrict__ WhT) {
  const int lane   = threadIdx.x & 31;
  const int wave   = threadIdx.x >> 5;     // 0..3
  const int mtile  = blockIdx.x;           // 0..1023
  const int r      = lane & 15;
  const int khalf  = (lane >> 4) * 16;     // A-fragment K mapping
  const int kgroup = (lane >> 4) * 16;     // B-fragment K mapping

  v8f acc[4] = {};

  for (int kt = 0; kt < FD / 32; ++kt) {
    // A fragment: row = mtile*16 + r, K = kt*32 + khalf + {0..15}
    const float* hrow = h + (size_t)(mtile * 16 + r) * FD + kt * 32 + khalf;
    const v4f* hp = reinterpret_cast<const v4f*>(hrow);
    const v4f x0 = hp[0], x1 = hp[1], x2 = hp[2], x3 = hp[3];
    v16h afr;
#pragma unroll
    for (int u = 0; u < 4; ++u) {
      afr[u]      = (half_t)x0[u];
      afr[4 + u]  = (half_t)x1[u];
      afr[8 + u]  = (half_t)x2[u];
      afr[12 + u] = (half_t)x3[u];
    }

#pragma unroll
    for (int t = 0; t < 4; ++t) {
      const int n = (wave * 4 + t) * 16 + r;
      // B fragment: K = kt*32 + kgroup + u (u=0..15), contiguous in WT16[n][*]
      const v16h bfr = *reinterpret_cast<const v16h*>(
          WT16 + (size_t)n * FD + kt * 32 + kgroup);
      acc[t] = __builtin_amdgcn_wmma_f32_16x16x32_f16(
          false, afr, false, bfr, (short)0, acc[t], false, false);
    }
  }

  // Store transposed: C layout -> lane holds column n, rows rbase..rbase+7
  const int rbase = (lane < 16) ? 0 : 8;
#pragma unroll
  for (int t = 0; t < 4; ++t) {
    const int n = (wave * 4 + t) * 16 + r;
    v8h st;
#pragma unroll
    for (int v = 0; v < 8; ++v) st[v] = (half_t)acc[t][v];
    *reinterpret_cast<v8h*>(WhT + (size_t)n * MM + mtile * 16 + rbase) = st;
  }
}

// ---------------------------------------------------------------------------
// Kernel 3: fused masked-softmax attention + attn@Wh + ELU (flash style).
// Grid: B * N/16 = 1024 blocks x 128 threads (4 waves).
// Block = (batch b, 16-row tile); wave w owns features [64w, 64w+64)
// (4 WMMA accumulators per wave -> softmax VALU amortized over 4 WMMAs).
// ---------------------------------------------------------------------------
__global__ __launch_bounds__(128)
void k_attn(const half_t* __restrict__ WhT, const int* __restrict__ adj,
            const float* __restrict__ e_i, const float* __restrict__ e_j,
            float* __restrict__ out) {
  const int lane   = threadIdx.x & 31;
  const int wave   = threadIdx.x >> 5;    // 0..3
  const int b      = blockIdx.x >> 6;     // 0..15
  const int itile  = blockIdx.x & 63;     // 0..63
  const int r      = lane & 15;
  const int khalf  = (lane >> 4) * 16;    // A-fragment K mapping
  const int kgroup = (lane >> 4) * 16;    // B-fragment K mapping
  const int rbase  = (lane < 16) ? 0 : 8;

  const int    i    = itile * 16 + r;     // this lane's row within batch
  const size_t rowb = (size_t)b * NN;

  const float ei = e_i[rowb + i];
  const half_t* wp[4];
#pragma unroll
  for (int t = 0; t < 4; ++t)
    wp[t] = WhT + (size_t)(wave * 64 + t * 16 + r) * MM + rowb;

  const int*   adjr = adj + ((size_t)b * NN + i) * NN;
  const float* ejr  = e_j + rowb;

  float m_run = -3.0e38f;
  float l_run = 0.f;
  v8f acc[4] = {};

  for (int jt = 0; jt < NN / 32; ++jt) {
    const int jb = jt * 32;

    // Gather e_j / adj for this lane's A-layout K set: khalf+{0..15}
    const v4f ej0 = *reinterpret_cast<const v4f*>(ejr + jb + khalf);
    const v4f ej1 = *reinterpret_cast<const v4f*>(ejr + jb + khalf + 4);
    const v4f ej2 = *reinterpret_cast<const v4f*>(ejr + jb + khalf + 8);
    const v4f ej3 = *reinterpret_cast<const v4f*>(ejr + jb + khalf + 12);
    const v4i ad0 = __builtin_nontemporal_load(
        reinterpret_cast<const v4i*>(adjr + jb + khalf));
    const v4i ad1 = __builtin_nontemporal_load(
        reinterpret_cast<const v4i*>(adjr + jb + khalf + 4));
    const v4i ad2 = __builtin_nontemporal_load(
        reinterpret_cast<const v4i*>(adjr + jb + khalf + 8));
    const v4i ad3 = __builtin_nontemporal_load(
        reinterpret_cast<const v4i*>(adjr + jb + khalf + 12));

    float ejv[16];
    int   adv[16];
#pragma unroll
    for (int u = 0; u < 4; ++u) {
      ejv[u] = ej0[u]; ejv[4 + u] = ej1[u]; ejv[8 + u] = ej2[u]; ejv[12 + u] = ej3[u];
      adv[u] = ad0[u]; adv[4 + u] = ad1[u]; adv[8 + u] = ad2[u]; adv[12 + u] = ad3[u];
    }

    float s[16];
    float mloc = -3.0e38f;
#pragma unroll
    for (int u = 0; u < 16; ++u) {
      float ev = ei + ejv[u];
      ev = (ev > 0.f) ? ev : (LRELU_ALPHA * ev);          // leaky relu
      ev = (adv[u] > 0) ? ev : NEG_BIG;                   // adjacency mask
      s[u] = ev;
      mloc = fmaxf(mloc, ev);
    }
    // combine partner lane (same row, other K half) -> row max of this tile
    mloc = fmaxf(mloc, __shfl_xor(mloc, 16, 32));

    const float m_new = fmaxf(m_run, mloc);
    const float scale = __expf(m_run - m_new);
    l_run *= scale;

    v16h pa;
    float ls = 0.f;
#pragma unroll
    for (int u = 0; u < 16; ++u) {
      const float p = __expf(s[u] - m_new);
      ls += p;
      pa[u] = (half_t)p;
    }
    l_run += ls;
    m_run = m_new;

    // rescale accumulators: per-row factors broadcast from lanes rbase+v
#pragma unroll
    for (int v = 0; v < 8; ++v) {
      const float rs = __shfl(scale, rbase + v, 32);
#pragma unroll
      for (int t = 0; t < 4; ++t) acc[t][v] *= rs;
    }

    // B fragments: contiguous 16 halfs (32B) per lane from transposed Wh
#pragma unroll
    for (int t = 0; t < 4; ++t) {
      const v16h bf = *reinterpret_cast<const v16h*>(wp[t] + jb + kgroup);
      acc[t] = __builtin_amdgcn_wmma_f32_16x16x32_f16(
          false, pa, false, bf, (short)0, acc[t], false, false);
    }
  }

  // finalize: combine partner partial sums, normalize, ELU, store (streaming)
  const float l_tot = l_run + __shfl_xor(l_run, 16, 32);
  const float linv  = 1.f / l_tot;

  float* orow = out + ((size_t)b * NN + itile * 16) * FD;
#pragma unroll
  for (int v = 0; v < 8; ++v) {
    const float rs = __shfl(linv, rbase + v, 32);
#pragma unroll
    for (int t = 0; t < 4; ++t) {
      float o = acc[t][v] * rs;
      o = (o > 0.f) ? o : (__expf(o) - 1.f);              // ELU
      __builtin_nontemporal_store(
          o, orow + (size_t)(rbase + v) * FD + wave * 64 + t * 16 + r);
    }
  }
}

// ---------------------------------------------------------------------------
extern "C" void kernel_launch(void* const* d_in, const int* in_sizes, int n_in,
                              void* d_out, int out_size, void* d_ws, size_t ws_size,
                              hipStream_t stream) {
  (void)in_sizes; (void)n_in; (void)out_size; (void)ws_size;

  const float* h   = (const float*)d_in[0];   // (16,1024,256) f32
  const int*   adj = (const int*)  d_in[1];   // (16,1024,1024) i32
  const float* W   = (const float*)d_in[2];   // (256,256) f32
  const float* a   = (const float*)d_in[3];   // (512,1) f32
  float*       out = (float*)d_out;           // (16,1024,256) f32

  // Workspace layout
  half_t* WhT  = (half_t*)d_ws;                                 // 8 MB
  float*  e_i  = (float*)((char*)d_ws + (size_t)FD * MM * sizeof(half_t));
  float*  e_j  = e_i + MM;                                      // 64 KB each
  float*  wa   = e_j + MM;                                      // 2 KB
  half_t* WT16 = (half_t*)(wa + 2 * FD);                        // 128 KB

  k_wt    <<<FD,             256, 0, stream>>>(W, WT16);
  k_wa    <<<1,              256, 0, stream>>>(W, a, wa);
  k_scores<<<MM / 8,         256, 0, stream>>>(h, wa, e_i, e_j);
  k_gemm1 <<<MM / 16,        128, 0, stream>>>(h, WT16, WhT);
  k_attn  <<<BB * (NN / 16), 128, 0, stream>>>(WhT, adj, e_i, e_j, out);
}
